// DepthRefinement_p1_231928234438
// MI455X (gfx1250) — matrix operand order repaired and checked by reading, not compiled, not run on hardware
//
#include <hip/hip_runtime.h>
#include <hip/hip_bf16.h>

typedef __attribute__((ext_vector_type(16))) _Float16 v16h;
typedef __attribute__((ext_vector_type(8)))  float    v8f;

#define HDIM 256
#define WDIM 320
#define HWSZ (HDIM * WDIM)

// ---------------------------------------------------------------------------
// Utility kernels
// ---------------------------------------------------------------------------
__global__ void fill_f_kernel(float* p, float v, int n) {
  int i = blockIdx.x * blockDim.x + threadIdx.x;
  if (i < n) p[i] = v;
}
__global__ void fill_h_kernel(_Float16* p, int n) {
  int i = blockIdx.x * blockDim.x + threadIdx.x;
  if (i < n) p[i] = (_Float16)0.0f;
}
__global__ void copy_bias_kernel(const float* __restrict__ src, float* __restrict__ dst,
                                 int n, int off) {
  int i = blockIdx.x * blockDim.x + threadIdx.x;
  if (i < n) dst[off + i] = src[i];
}

// ---------------------------------------------------------------------------
// Weight packing: OIHW fp32 (+ folded bn scale) -> f16 WMMA B fragments.
// Fragment layout (per CDNA5 ISA 16-bit B striping):
//   lane L holds column N = L&15; within a 32-wide K chunk, lane's 16 halves
//   cover K = (L>>4)*8 + (hi&7) + (hi>>3)*16.  Each lane's 16 halves stored contiguously:
//   Bp[((kc*NT + nt)*32 + lane)*16 + hi]  -> one coalesced 32B load per lane.
// ---------------------------------------------------------------------------
__global__ void pack_B_kernel(const float* __restrict__ W, const float* __restrict__ scale,
                              _Float16* __restrict__ Bp, int O, int Kreal, int n_base, int NT) {
  int idx = blockIdx.x * blockDim.x + threadIdx.x;
  int total = O * Kreal;
  if (idx >= total) return;
  int o = idx / Kreal;
  int k = idx - o * Kreal;           // k = ci*KS*KS + kh*KS + kw (matches A build)
  float v = W[idx];
  if (scale) v *= scale[o];
  int n = n_base + o;
  int kc = k >> 5, rem = k & 31;
  int laneHi = (rem >> 3) & 1, hi = (rem & 7) | ((rem >> 4) << 3);
  int nt = n >> 4, laneLo = n & 15;
  int lane = (laneHi << 4) | laneLo;
  Bp[((((size_t)kc * NT + nt) * 32 + lane) << 4) + hi] = (_Float16)v;
}

// ---------------------------------------------------------------------------
// Generic conv-as-GEMM with WMMA f16 (f32 accumulate).
// One wave computes a 16-pixel x (NT*16)-outchannel tile.
// The K loop is FULLY UNROLLED so every im2col index (ci, dh, dw) constant-
// folds: A-element = one predicated load at a literal offset from a per-lane
// base pointer (no runtime div/mod).
// A fragment per ISA 16-bit A layout: lane L -> row M = L&15,
//   half hi -> K = (L>>4)*8 + (hi&7) + (hi>>3)*16 within the chunk.
// W=320 is a multiple of 16, so a 16-pixel tile never crosses a row.
// ---------------------------------------------------------------------------
template <int CIN, int KS, int NT, bool RELU, bool HASADD, bool HASSCALE>
__global__ __launch_bounds__(256)
void conv_wmma_kernel(const float* __restrict__ in, const float* __restrict__ addend,
                      const float* __restrict__ inscale, const _Float16* __restrict__ Bp,
                      const float* __restrict__ bias, float* __restrict__ out, int Cout) {
  constexpr int P = (KS - 1) / 2;
  constexpr int KREAL = CIN * KS * KS;
  constexpr int KC = (KREAL + 31) / 32;
  constexpr int MTILES = HWSZ / 16;

  int wave = blockIdx.x * (blockDim.x >> 5) + (threadIdx.x >> 5);
  if (wave >= MTILES) return;
  int lane = threadIdx.x & 31;
  int base = wave << 4;
  int m = lane & 15;
  int p = base + m;
  int h = p / WDIM;
  int w = p - h * WDIM;
  int kh0 = (lane >> 4) * 8;

  const float* __restrict__ pix  = in + (size_t)h * WDIM + w;
  const float* __restrict__ apix = HASADD ? (addend + (size_t)h * WDIM + w) : nullptr;
  const _Float16* __restrict__ bbase = Bp + ((size_t)lane << 4);

  v8f acc[NT];
#pragma unroll
  for (int nt = 0; nt < NT; ++nt) {
#pragma unroll
    for (int r = 0; r < 8; ++r) acc[nt][r] = 0.0f;
  }

#pragma unroll
  for (int kc = 0; kc < KC; ++kc) {
    // ---- build A fragment (im2col on the fly, fp32 -> f16) ----
    v16h a;
#pragma unroll
    for (int hi = 0; hi < 16; ++hi) {
      const int k = kc * 32 + kh0 + (hi & 7) + ((hi >> 3) << 4);  // kh0 runtime 0/8
      float v = 0.0f;
      // two compile-time variants (kh0 = 0 or 8) resolved by the runtime term;
      // ci/dh/dw below still constant-fold per variant after unrolling.
      if (k < KREAL) {
        const int ci = k / (KS * KS);
        const int r  = k - ci * (KS * KS);
        const int dh = r / KS - P;
        const int dw = r % KS - P;
        if ((unsigned)(h + dh) < (unsigned)HDIM && (unsigned)(w + dw) < (unsigned)WDIM) {
          const ptrdiff_t ofs = (ptrdiff_t)ci * HWSZ + (ptrdiff_t)dh * WDIM + dw;
          v = pix[ofs];
          if (HASSCALE) v *= inscale[ci];
          if (HASADD) v += apix[ofs];
        }
      }
      a[hi] = (_Float16)v;
    }
    if (kc + 1 < KC)
      __builtin_prefetch(bbase + (size_t)(kc + 1) * NT * 512, 0, 1);
    // ---- multiply against all N tiles, reusing A ----
#pragma unroll
    for (int nt = 0; nt < NT; ++nt) {
      v16h b = *(const v16h*)(bbase + (size_t)(kc * NT + nt) * 512);
      acc[nt] = __builtin_amdgcn_wmma_f32_16x16x32_f16(
          false, a, false, b, (short)0, acc[nt], false, false);
    }
  }

  // ---- epilogue: D layout lane L -> N=L&15, VGPR r -> M = r + 8*(L>>4) ----
  int mrow0 = (lane >> 4) * 8;
#pragma unroll
  for (int nt = 0; nt < NT; ++nt) {
    int n = nt * 16 + (lane & 15);
    if (n >= Cout) continue;
    float bs = bias ? bias[n] : 0.0f;
#pragma unroll
    for (int r = 0; r < 8; ++r) {
      float v = acc[nt][r] + bs;
      if (RELU) v = fmaxf(v, 0.0f);
      out[(size_t)n * HWSZ + base + mrow0 + r] = v;
    }
  }
}

// ---------------------------------------------------------------------------
// SE block: channel means (16 ch) -> fc16 relu -> fc16 sigmoid
// ---------------------------------------------------------------------------
__global__ void se_reduce_kernel(const float* __restrict__ xy, float* __restrict__ zsum) {
  __shared__ float red[256];
  int c = blockIdx.x;
  float s = 0.0f;
  for (int p = threadIdx.x; p < HWSZ; p += 256) s += xy[(size_t)c * HWSZ + p];
  red[threadIdx.x] = s;
  __syncthreads();
  for (int st = 128; st > 0; st >>= 1) {
    if (threadIdx.x < st) red[threadIdx.x] += red[threadIdx.x + st];
    __syncthreads();
  }
  if (threadIdx.x == 0) zsum[c] = red[0];
}

__global__ void se_fc_kernel(const float* __restrict__ zsum, const float* __restrict__ W1,
                             const float* __restrict__ W2, float* __restrict__ z) {
  __shared__ float z0[16], z1[16];
  int t = threadIdx.x;
  if (t < 16) z0[t] = zsum[t] * (1.0f / (float)HWSZ);
  __syncthreads();
  if (t < 16) {
    float s = 0.0f;
    for (int k = 0; k < 16; ++k) s += z0[k] * W1[t * 16 + k];
    z1[t] = fmaxf(s, 0.0f);
  }
  __syncthreads();
  if (t < 16) {
    float s = 0.0f;
    for (int k = 0; k < 16; ++k) s += z1[k] * W2[t * 16 + k];
    z[t] = 1.0f / (1.0f + expf(-s));
  }
}

// ---------------------------------------------------------------------------
// Heads post-processing: mask sigmoid*valid, kconf softmax, iconf softmaxes,
// per-pixel mix weights, and L1-normalized CSPN guides (encoder conv is a pure
// spatial shuffle and is folded into the CSPN step instead).
// raw channel map: mask 0..2, kconf 3..5, iconf 6..17, g3 18..25, g5 26..49, g7 50..97
// ---------------------------------------------------------------------------
__device__ __forceinline__ void guide_norm(const float* __restrict__ g, int Ksub,
                                           float* __restrict__ outg, int p) {
  float s = 0.0f;
  for (int c = 0; c < Ksub; ++c) s += fabsf(g[(size_t)c * HWSZ + p]);
  float inv = 1.0f / s;
  int half = Ksub >> 1;
  float tot = 0.0f;
  for (int c = 0; c < Ksub; ++c) {
    float gn = g[(size_t)c * HWSZ + p] * inv;
    tot += gn;
    int oc = (c < half) ? c : c + 1;
    outg[(size_t)oc * HWSZ + p] = gn;
  }
  outg[(size_t)half * HWSZ + p] = 1.0f - tot;
}

__global__ void heads_post_kernel(const float* __restrict__ raw, const float* __restrict__ sparse,
                                  float* __restrict__ mask, float* __restrict__ wts,
                                  float* __restrict__ g3, float* __restrict__ g5,
                                  float* __restrict__ g7) {
  int p = blockIdx.x * blockDim.x + threadIdx.x;
  if (p >= HWSZ) return;
  float valid = sparse[p] > 0.0f ? 1.0f : 0.0f;
#pragma unroll
  for (int c = 0; c < 3; ++c)
    mask[(size_t)c * HWSZ + p] = valid / (1.0f + expf(-raw[(size_t)c * HWSZ + p]));
  // kconf softmax (3)
  float kv[3], km = -1e30f;
#pragma unroll
  for (int c = 0; c < 3; ++c) { kv[c] = raw[(size_t)(3 + c) * HWSZ + p]; km = fmaxf(km, kv[c]); }
  float ksum = 0.0f;
#pragma unroll
  for (int c = 0; c < 3; ++c) { kv[c] = expf(kv[c] - km); ksum += kv[c]; }
#pragma unroll
  for (int c = 0; c < 3; ++c) kv[c] /= ksum;
  // iconf: three softmax groups of 4; wts[(t*4+s)] = kconf_t * conf_t[s]
#pragma unroll
  for (int t = 0; t < 3; ++t) {
    float cv[4], cm = -1e30f;
#pragma unroll
    for (int s = 0; s < 4; ++s) { cv[s] = raw[(size_t)(6 + 4 * t + s) * HWSZ + p]; cm = fmaxf(cm, cv[s]); }
    float cs = 0.0f;
#pragma unroll
    for (int s = 0; s < 4; ++s) { cv[s] = expf(cv[s] - cm); cs += cv[s]; }
#pragma unroll
    for (int s = 0; s < 4; ++s) wts[(size_t)(t * 4 + s) * HWSZ + p] = kv[t] * (cv[s] / cs);
  }
  guide_norm(raw + (size_t)18 * HWSZ, 8, g3, p);
  guide_norm(raw + (size_t)26 * HWSZ, 24, g5, p);
  guide_norm(raw + (size_t)50 * HWSZ, 48, g7, p);
}

// ---------------------------------------------------------------------------
// CSPN iteration (fp32, stencil). Encoder shuffle folded in:
//   d_new[h,w] = sum_j garr[j, h-dy, w-dx] * (j==mid ? pred[h,w] : d[h-dy, w-dx])
// with dy = j/ks - P, dx = j%ks - P; OOB terms vanish (zero-pad on both sides).
// ---------------------------------------------------------------------------
template <int KSZ>
__device__ __forceinline__ float cspn_apply(const float* __restrict__ g,
                                            const float* __restrict__ d, float pr, int h, int w) {
  constexpr int P = (KSZ - 1) / 2, K = KSZ * KSZ, MID = K / 2;
  float acc = 0.0f;
#pragma unroll
  for (int j = 0; j < K; ++j) {
    int hh = h - (j / KSZ - P);
    int ww = w - (j % KSZ - P);
    if ((unsigned)hh < (unsigned)HDIM && (unsigned)ww < (unsigned)WDIM) {
      float gv = g[(size_t)j * HWSZ + (size_t)hh * WDIM + ww];
      float dv = (j == MID) ? pr : d[(size_t)hh * WDIM + ww];
      acc += gv * dv;
    }
  }
  return acc;
}

__global__ void init_d_kernel(const float* __restrict__ pred, float* __restrict__ d,
                              float* __restrict__ out) {
  int p = blockIdx.x * blockDim.x + threadIdx.x;
  if (p >= HWSZ) return;
  float v = pred[p];
  d[p] = v; d[HWSZ + p] = v; d[2 * HWSZ + p] = v;
  out[p] = 0.0f;
}

__global__ void cspn_step_kernel(const float* __restrict__ g3, const float* __restrict__ g5,
                                 const float* __restrict__ g7, const float* __restrict__ mask,
                                 const float* __restrict__ wts, const float* __restrict__ sparse,
                                 const float* __restrict__ pred, const float* __restrict__ din,
                                 float* __restrict__ dout, float* __restrict__ out, int snap) {
  int p = blockIdx.x * blockDim.x + threadIdx.x;
  if (p >= HWSZ) return;
  int h = p / WDIM, w = p - h * WDIM;
  float pr = pred[p], sp = sparse[p];
  float d0 = cspn_apply<3>(g3, din, pr, h, w);
  float d1 = cspn_apply<5>(g5, din + HWSZ, pr, h, w);
  float d2 = cspn_apply<7>(g7, din + 2 * HWSZ, pr, h, w);
  float m0 = mask[p], m1 = mask[HWSZ + p], m2 = mask[2 * HWSZ + p];
  d0 = m0 * sp + (1.0f - m0) * d0;
  d1 = m1 * sp + (1.0f - m1) * d1;
  d2 = m2 * sp + (1.0f - m2) * d2;
  dout[p] = d0; dout[HWSZ + p] = d1; dout[2 * HWSZ + p] = d2;
  if (snap >= 0)
    out[p] += wts[(size_t)snap * HWSZ + p] * d0 +
              wts[(size_t)(4 + snap) * HWSZ + p] * d1 +
              wts[(size_t)(8 + snap) * HWSZ + p] * d2;
}

// ---------------------------------------------------------------------------
// Host orchestration
// ---------------------------------------------------------------------------
static inline int divup(int a, int b) { return (a + b - 1) / b; }

extern "C" void kernel_launch(void* const* d_in, const int* in_sizes, int n_in,
                              void* d_out, int out_size, void* d_ws, size_t ws_size,
                              hipStream_t stream) {
  (void)in_sizes; (void)n_in; (void)out_size; (void)ws_size;
  const float* rgb    = (const float*)d_in[1];
  const float* y      = (const float*)d_in[3];
  const float* sparse = (const float*)d_in[5];
  const float* pred   = (const float*)d_in[6];
  // params flattened as JAX pytree (sorted dict keys). Unused rch/sch skipped.
  enum { G3 = 7, G5 = 10, G7 = 13, ICONF = 16, KCONF = 19, MASK = 22, RC = 25, SC = 69 };
  enum { cW1 = 0, cW2, cb1, cb2, cs1, cs2, fW1, fW2, fb1, fb2, fs1, fs2,
         gW1, gW2, gb1, gb2, gs1, gs2, rW, rb, seW1, seW2 };
  auto F = [&](int i) { return (const float*)d_in[i]; };

  // workspace layout (everything fits in L2: ~100MB)
  float* fw = (float*)d_ws;
  size_t off = 0;
  auto FA = [&](size_t n) { float* r = fw + off; off += n; return r; };
  float* bufA  = FA((size_t)16 * HWSZ);
  float* bufB  = FA((size_t)16 * HWSZ);
  float* feat  = FA((size_t)64 * HWSZ);
  float* raw   = FA((size_t)98 * HWSZ);
  float* maskb = FA((size_t)3 * HWSZ);
  float* wtsb  = FA((size_t)12 * HWSZ);
  float* g3b   = FA((size_t)9 * HWSZ);
  float* g5b   = FA((size_t)25 * HWSZ);
  float* g7b   = FA((size_t)49 * HWSZ);
  float* dping = FA((size_t)3 * HWSZ);
  float* dpong = FA((size_t)3 * HWSZ);
  float* zsum  = FA(16);
  float* zvec  = FA(16);
  float* biasH = FA(128);
  _Float16* slot    = (_Float16*)(fw + off);  // reusable single-conv B slot
  _Float16* headsBp = slot + 4096;            // 18 chunks * 7 ntiles * 512

  const int CBLK = (HWSZ / 16) / 8;  // 640 blocks of 8 waves
  const int PBLK = divup(HWSZ, 256);

  auto packB = [&](const float* Wt, const float* sc, _Float16* Bp, int O, int Kr, int NT) {
    int KC = divup(Kr, 32);
    int nB = KC * NT * 512;
    fill_h_kernel<<<divup(nB, 256), 256, 0, stream>>>(Bp, nB);
    pack_B_kernel<<<divup(O * Kr, 256), 256, 0, stream>>>(Wt, sc, Bp, O, Kr, 0, NT);
  };

  auto run_gub = [&](int base, const float* guide, int gch, float* feat_out) {
    auto P = [&](int i) { return (const float*)d_in[base + i]; };
    packB(P(fW1), P(fs1), slot, 16, 16 * 9, 1);
    conv_wmma_kernel<16, 3, 1, true, false, false><<<CBLK, 256, 0, stream>>>(
        y, nullptr, nullptr, slot, P(fb1), bufA, 16);
    packB(P(fW2), P(fs2), slot, 8, 16, 1);
    conv_wmma_kernel<16, 1, 1, true, false, false><<<CBLK, 256, 0, stream>>>(
        bufA, nullptr, nullptr, slot, P(fb2), bufB, 8);
    packB(P(gW1), P(gs1), slot, 16, gch * 9, 1);
    if (gch == 3)
      conv_wmma_kernel<3, 3, 1, true, false, false><<<CBLK, 256, 0, stream>>>(
          guide, nullptr, nullptr, slot, P(gb1), bufA, 16);
    else
      conv_wmma_kernel<1, 3, 1, true, false, false><<<CBLK, 256, 0, stream>>>(
          guide, nullptr, nullptr, slot, P(gb1), bufA, 16);
    packB(P(gW2), P(gs2), slot, 8, 16, 1);
    conv_wmma_kernel<16, 1, 1, true, false, false><<<CBLK, 256, 0, stream>>>(
        bufA, nullptr, nullptr, slot, P(gb2), bufB + (size_t)8 * HWSZ, 8);
    se_reduce_kernel<<<16, 256, 0, stream>>>(bufB, zsum);
    se_fc_kernel<<<1, 32, 0, stream>>>(zsum, P(seW1), P(seW2), zvec);
    packB(P(cW1), P(cs1), slot, 16, 16 * 9, 1);
    conv_wmma_kernel<16, 3, 1, true, false, true><<<CBLK, 256, 0, stream>>>(
        bufB, nullptr, zvec, slot, P(cb1), bufA, 16);
    packB(P(cW2), P(cs2), slot, 16, 16, 1);
    conv_wmma_kernel<16, 1, 1, true, false, false><<<CBLK, 256, 0, stream>>>(
        bufA, nullptr, nullptr, slot, P(cb2), bufB, 16);
    packB(P(rW), nullptr, slot, 32, 16, 2);
    conv_wmma_kernel<16, 1, 2, false, true, false><<<CBLK, 256, 0, stream>>>(
        bufB, y, nullptr, slot, P(rb), feat_out, 32);
  };

  run_gub(RC, rgb, 3, feat);
  run_gub(SC, sparse, 1, feat + (size_t)32 * HWSZ);

  // fused heads: 3x3 conv 64 -> 98 (mask|kconf|iconf|g3|g5|g7), bn folded
  {
    int nB = 18 * 7 * 512;
    fill_h_kernel<<<divup(nB, 256), 256, 0, stream>>>(headsBp, nB);
    fill_f_kernel<<<1, 128, 0, stream>>>(biasH, 0.0f, 128);
    const int hb[6]   = {MASK, KCONF, ICONF, G3, G5, G7};
    const int hO[6]   = {3, 3, 12, 8, 24, 48};
    const int hoff[6] = {0, 3, 6, 18, 26, 50};
    for (int i = 0; i < 6; ++i) {
      pack_B_kernel<<<divup(hO[i] * 576, 256), 256, 0, stream>>>(
          F(hb[i] + 0), F(hb[i] + 2), headsBp, hO[i], 576, hoff[i], 7);
      copy_bias_kernel<<<1, 64, 0, stream>>>(F(hb[i] + 1), biasH, hO[i], hoff[i]);
    }
    conv_wmma_kernel<64, 3, 7, false, false, false><<<CBLK, 256, 0, stream>>>(
        feat, nullptr, nullptr, headsBp, biasH, raw, 98);
  }

  heads_post_kernel<<<PBLK, 256, 0, stream>>>(raw, sparse, maskb, wtsb, g3b, g5b, g7b);
  init_d_kernel<<<PBLK, 256, 0, stream>>>(pred, dping, (float*)d_out);

  float* dcur = dping;
  float* dnext = dpong;
  for (int it = 0; it < 12; ++it) {
    int snap = (it % 3 == 2) ? it / 3 : -1;  // iterations 2,5,8,11 -> snapshots 0..3
    cspn_step_kernel<<<PBLK, 256, 0, stream>>>(g3b, g5b, g7b, maskb, wtsb, sparse, pred,
                                               dcur, dnext, (float*)d_out, snap);
    float* t = dcur; dcur = dnext; dnext = t;
  }
}
